// MHSA_8409545965580
// MI455X (gfx1250) — matrix operand order repaired and checked by reading, not compiled, not run on hardware
//
#include <hip/hip_runtime.h>

// MHSA fwd: B=2, S=2048, D=1024, H=16, HD=64.
// All matmuls on v_wmma_f32_16x16x32_bf16; attention K/V tiles staged with the
// Tensor Data Mover (TENSOR_LOAD_TO_LDS) and double-buffered via TENSORcnt.
#define BB 2
#define SS 2048
#define DD 1024
#define HH 16
#define HD 64
#define MM (BB * SS)  // 4096 total rows

typedef __bf16 bf16_t;
typedef __attribute__((ext_vector_type(16))) __bf16 v16bf;
typedef __attribute__((ext_vector_type(8)))  float  v8f;
typedef unsigned int u32;
typedef __attribute__((ext_vector_type(4))) unsigned int u32x4;
typedef __attribute__((ext_vector_type(8))) int i32x8;
typedef __attribute__((ext_vector_type(4))) int i32x4;

// 16 bf16 = 32B fragment; union lets us do 16B vector loads and per-element converts.
union Frag16 { v16bf v; bf16_t h[16]; uint4 u[2]; };

__device__ __forceinline__ v8f wmma_bf16(v16bf a, v16bf b, v8f c) {
  // (neg_a, A, neg_b, B, c_mod, C, reuse_a, reuse_b)
  return __builtin_amdgcn_wmma_f32_16x16x32_bf16(false, a, false, b, (short)0, c, false, false);
}

// ---------------------------------------------------------------------------
// TDM: load a 2D tile (tile1 rows x tile0 elems, 2-byte elements) from global
// (row stride stride0 elems) into LDS at lds_off, inserting 4 DWORDs (16B) of
// LDS padding after every 32 DWORDs (128B = one 64-elem bf16 row) -> LDS row
// stride 72 bf16.  D# layout per CDNA5 ISA.
// ---------------------------------------------------------------------------
__device__ __forceinline__ void tdm_load_2d(const void* gptr, u32 lds_off,
                                            u32 tdim0, u32 tdim1, u32 stride0,
                                            u32 tile0, u32 tile1) {
  const unsigned long long ga = (unsigned long long)gptr;
  u32x4 g0;
  g0.x = 1u;                              // count=1, user descriptor
  g0.y = lds_off;                         // lds_addr [63:32]
  g0.z = (u32)ga;                         // global_addr [95:64]
  g0.w = (u32)(ga >> 32) | (2u << 30);    // global_addr hi + type=2 ("image")
  i32x8 g1;
  g1[0] = (int)((1u << 16)                // data_size = 1 -> 2 bytes
                | (1u << 20)              // pad_enable
                | (4u << 22)              // pad_interval: 32 DWORDs (128B)
                | (3u << 25));            // pad_amount: 4 DWORDs (16B)
  g1[1] = (int)((tdim0 & 0xFFFFu) << 16);                 // tensor_dim0 lo
  g1[2] = (int)((tdim0 >> 16) | ((tdim1 & 0xFFFFu) << 16));
  g1[3] = (int)((tdim1 >> 16) | (tile0 << 16));           // tile_dim0
  g1[4] = (int)(tile1 & 0xFFFFu);                         // tile_dim1 (dim2=0)
  g1[5] = (int)stride0;                                   // tensor_dim0_stride
  g1[6] = 0;
  g1[7] = 0;
  const i32x4 z4 = {0, 0, 0, 0};
#if __clang_major__ >= 23
  const i32x8 z8 = {0, 0, 0, 0, 0, 0, 0, 0};
  __builtin_amdgcn_tensor_load_to_lds(g0, g1, z4, z4, z8, 0);
#else
  __builtin_amdgcn_tensor_load_to_lds(g0, g1, z4, z4, 0);
#endif
}

// -------------------------------------------------------------------------
// Kernel 1: fused QKV projection.  C[m,n] = sum_k x[m,k] * w[n,k]  (w is [out,in])
//   grid = (D/64, M/128, 3)   block = 256 (8 waves, 16 rows each, 64 cols)
//   N-block of 64 == exactly one head. Q scaled by 1/sqrt(64); V stored transposed.
// -------------------------------------------------------------------------
__global__ __launch_bounds__(256) void qkv_proj_kernel(
    const float* __restrict__ x, const float* __restrict__ qw,
    const float* __restrict__ kw, const float* __restrict__ vw,
    bf16_t* __restrict__ qb, bf16_t* __restrict__ kb, bf16_t* __restrict__ vtb) {
  __shared__ __align__(16) bf16_t wt[64][40];  // 64 N-rows x 32 K (+8 pad)

  const int nBase = blockIdx.x * 64;   // == head * 64
  const int mBase = blockIdx.y * 128;
  const int mode  = blockIdx.z;        // 0=Q 1=K 2=V
  const float* w = (mode == 0) ? qw : ((mode == 1) ? kw : vw);

  const int tid = threadIdx.x, lane = tid & 31, wave = tid >> 5;
  const int mWave = mBase + wave * 16;
  const int l15 = lane & 15, khalf = lane >> 4;

  v8f acc[4] = {};
  const float* xrow = x + (size_t)(mWave + l15) * DD;

  for (int k0 = 0; k0 < DD; k0 += 32) {
    __syncthreads();
    {  // stage weight tile fp32 -> bf16 into LDS (shared by all waves)
      const int r = tid >> 2, c = (tid & 3) * 8;
      const float* src = w + (size_t)(nBase + r) * DD + k0 + c;
      bf16_t* dst = &wt[r][c];
#pragma unroll
      for (int j = 0; j < 8; ++j) dst[j] = (bf16_t)src[j];
    }
    __syncthreads();

    // A fragment: lane holds row l15, K chunks at khalf*8 and khalf*8+16 (ISA layout)
    Frag16 a;
    const float* xr = xrow + k0 + khalf * 8;
#pragma unroll
    for (int j = 0; j < 8; ++j) { a.h[j] = (bf16_t)xr[j]; a.h[8 + j] = (bf16_t)xr[16 + j]; }

#pragma unroll
    for (int nf = 0; nf < 4; ++nf) {
      // B fragment: lane holds column N = nf*16+l15 -> row of [N,K] weight tile, contiguous
      Frag16 bfr;
      const bf16_t* bp = &wt[nf * 16 + l15][khalf * 8];
      bfr.u[0] = *(const uint4*)bp;
      bfr.u[1] = *(const uint4*)(bp + 16);
      acc[nf] = wmma_bf16(a.v, bfr.v, acc[nf]);
    }
  }

  // C layout: VGPR i -> row i (lanes 0-15) / row i+8 (lanes 16-31), col = l15
  const int h = blockIdx.x;
#pragma unroll
  for (int nf = 0; nf < 4; ++nf) {
#pragma unroll
    for (int i = 0; i < 8; ++i) {
      const int m = mWave + i + ((lane & 16) ? 8 : 0);
      const int hd = nf * 16 + l15;
      const int b = m >> 11, s = m & (SS - 1);
      const float vval = acc[nf][i];
      if (mode == 0)
        qb[(((size_t)(b * HH + h)) * SS + s) * HD + hd] = (bf16_t)(vval * 0.125f);
      else if (mode == 1)
        kb[(((size_t)(b * HH + h)) * SS + s) * HD + hd] = (bf16_t)vval;
      else
        vtb[(((size_t)(b * HH + h)) * HD + hd) * SS + s] = (bf16_t)vval;  // V transposed
    }
  }
}

// -------------------------------------------------------------------------
// Kernel 2: flash attention.  grid = (S/128, B*H)  block = 256 (8 waves x 16 q-rows)
// K/V^T tiles DMA'd by the TDM into double-buffered LDS, overlapped with WMMA.
// -------------------------------------------------------------------------
__global__ __launch_bounds__(256) void attn_kernel(
    const bf16_t* __restrict__ qb, const bf16_t* __restrict__ kb,
    const bf16_t* __restrict__ vtb, bf16_t* __restrict__ ob) {
  __shared__ __align__(16) bf16_t kt[2][64][72];   // K tile  [kv=64][hd=64] (+TDM pad)
  __shared__ __align__(16) bf16_t vt[2][64][72];   // V^T tile [hd=64][kv=64] (+TDM pad)
  __shared__ __align__(16) bf16_t pt[8][16][72];   // per-wave P repack (C->A layout)

  const int qBase = blockIdx.x * 128;
  const int bh = blockIdx.y;
  const int b = bh >> 4, h = bh & 15;
  const int tid = threadIdx.x, lane = tid & 31, wave = tid >> 5;
  const int qWave = qBase + wave * 16;
  const int l15 = lane & 15, khalf = lane >> 4;
  // Scalar wave id -> TDM issue guards become s_cbranch (TDM ignores EXEC, so a
  // lane-divergent guard would make every wave issue a duplicate DMA).
  const int waveU = __builtin_amdgcn_readfirstlane(tid) >> 5;

  const bf16_t* qhead = qb + (size_t)bh * SS * HD;
  const bf16_t* khead = kb + (size_t)bh * SS * HD;
  const bf16_t* vhead = vtb + (size_t)bh * HD * SS;

  // Q A-fragments (HD=64 -> two K-steps of 32), kept in registers for all kv tiles
  Frag16 qf[2];
  {
    const bf16_t* qrow = qhead + (size_t)(qWave + l15) * HD;
#pragma unroll
    for (int ks = 0; ks < 2; ++ks) {
      const bf16_t* p0 = qrow + ks * 32 + khalf * 8;
      qf[ks].u[0] = *(const uint4*)(p0);
      qf[ks].u[1] = *(const uint4*)(p0 + 16);
    }
  }

  v8f o[4] = {};
  float mrun[8], lrun[8];
#pragma unroll
  for (int i = 0; i < 8; ++i) { mrun[i] = -1e30f; lrun[i] = 0.0f; }

  const int nTiles = qBase / 64 + 2;  // causal: kv in [0, qBase+128)

  // Issue TDM loads of kv tile t into LDS buffer `buf` (wave0: K, wave1: V^T)
  auto issue_tiles = [&](int t, int buf) {
    const int kvBase = t * 64;
    if (waveU == 0)
      tdm_load_2d(khead + (size_t)kvBase * HD, (u32)(size_t)&kt[buf][0][0],
                  /*tdim0=*/HD, /*tdim1=*/SS, /*stride0=*/HD, 64, 64);
    else if (waveU == 1)
      tdm_load_2d(vhead + kvBase, (u32)(size_t)&vt[buf][0][0],
                  /*tdim0=*/SS, /*tdim1=*/HD, /*stride0=*/SS, 64, 64);
  };

  issue_tiles(0, 0);  // preload first tile

  for (int t = 0; t < nTiles; ++t) {
    const int cur = t & 1;
    const int kvBase = t * 64;

    // Wait for buf[cur] DMA (only issuing waves have TENSORcnt>0; no-op for rest),
    // then one barrier: publishes buf[cur] AND proves buf[cur^1] is done being read.
    __builtin_amdgcn_s_wait_tensorcnt(0);
    __syncthreads();
    if (t + 1 < nTiles) issue_tiles(t + 1, cur ^ 1);  // overlap next DMA with compute

    // scores S = Q K^T  (Q pre-scaled): 16 x 64 per wave, 8 WMMA
    v8f s[4] = {};
#pragma unroll
    for (int ks = 0; ks < 2; ++ks) {
#pragma unroll
      for (int nf = 0; nf < 4; ++nf) {
        Frag16 bfr;  // B col = kv row nf*16+l15 of K tile -> contiguous
        const bf16_t* bp = &kt[cur][nf * 16 + l15][ks * 32 + khalf * 8];
        bfr.u[0] = *(const uint4*)bp;
        bfr.u[1] = *(const uint4*)(bp + 16);
        s[nf] = wmma_bf16(qf[ks].v, bfr.v, s[nf]);
      }
    }

    // causal mask (elementwise, no WMMA under divergence)
    if (kvBase + 63 > qWave) {
#pragma unroll
      for (int nf = 0; nf < 4; ++nf) {
        const int col = kvBase + nf * 16 + l15;
#pragma unroll
        for (int i = 0; i < 8; ++i) {
          const int row = qWave + i + ((lane & 16) ? 8 : 0);
          if (col > row) s[nf][i] = -1e30f;
        }
      }
    }

    // online softmax: row r lives in (vgpr i, 16-lane half) -> xor-reduce masks 1..8
#pragma unroll
    for (int i = 0; i < 8; ++i) {
      float mv = fmaxf(fmaxf(s[0][i], s[1][i]), fmaxf(s[2][i], s[3][i]));
#pragma unroll
      for (int d = 1; d < 16; d <<= 1) mv = fmaxf(mv, __shfl_xor(mv, d, 32));
      const float mnew = fmaxf(mrun[i], mv);
      const float scl = __expf(mrun[i] - mnew);
      mrun[i] = mnew;
      float rs = 0.0f;
#pragma unroll
      for (int nf = 0; nf < 4; ++nf) {
        const float p = __expf(s[nf][i] - mnew);
        s[nf][i] = p;
        rs += p;
      }
#pragma unroll
      for (int d = 1; d < 16; d <<= 1) rs += __shfl_xor(rs, d, 32);
      lrun[i] = lrun[i] * scl + rs;
#pragma unroll
      for (int nf = 0; nf < 4; ++nf) o[nf][i] *= scl;
    }

    // repack P (C layout -> A layout) through per-wave LDS; same-wave DS ops are in-order
#pragma unroll
    for (int nf = 0; nf < 4; ++nf)
#pragma unroll
      for (int i = 0; i < 8; ++i)
        pt[wave][i + ((lane & 16) ? 8 : 0)][nf * 16 + l15] = (bf16_t)s[nf][i];

    // O += P V : 8 WMMA
#pragma unroll
    for (int ks = 0; ks < 2; ++ks) {
      Frag16 pa;
      const bf16_t* pp = &pt[wave][l15][ks * 32 + khalf * 8];
      pa.u[0] = *(const uint4*)pp;
      pa.u[1] = *(const uint4*)(pp + 16);
#pragma unroll
      for (int nf = 0; nf < 4; ++nf) {
        Frag16 vb;  // B col = hd row nf*16+l15 of V^T tile -> contiguous
        const bf16_t* vp = &vt[cur][nf * 16 + l15][ks * 32 + khalf * 8];
        vb.u[0] = *(const uint4*)vp;
        vb.u[1] = *(const uint4*)(vp + 16);
        o[nf] = wmma_bf16(pa.v, vb.v, o[nf]);
      }
    }
  }

  // normalize and store O as bf16 in [b, s, h*64+hd] so out-proj is a plain GEMM
#pragma unroll
  for (int i = 0; i < 8; ++i) {
    const float inv = 1.0f / lrun[i];
    const int row = qWave + i + ((lane & 16) ? 8 : 0);
#pragma unroll
    for (int nf = 0; nf < 4; ++nf)
      ob[((size_t)(b * SS + row)) * DD + h * HD + nf * 16 + l15] =
          (bf16_t)(o[nf][i] * inv);
  }
}

// -------------------------------------------------------------------------
// Kernel 3: output projection  out[m,n] = sum_k O[m,k] * o_w[n,k], fp32 out
// -------------------------------------------------------------------------
__global__ __launch_bounds__(256) void out_proj_kernel(
    const bf16_t* __restrict__ ob, const float* __restrict__ ow,
    float* __restrict__ out) {
  __shared__ __align__(16) bf16_t wt[64][40];

  const int nBase = blockIdx.x * 64;
  const int mBase = blockIdx.y * 128;
  const int tid = threadIdx.x, lane = tid & 31, wave = tid >> 5;
  const int mWave = mBase + wave * 16;
  const int l15 = lane & 15, khalf = lane >> 4;
  const bf16_t* arow = ob + (size_t)(mWave + l15) * DD;

  v8f acc[4] = {};
  for (int k0 = 0; k0 < DD; k0 += 32) {
    __syncthreads();
    {
      const int r = tid >> 2, c = (tid & 3) * 8;
      const float* src = ow + (size_t)(nBase + r) * DD + k0 + c;
      bf16_t* dst = &wt[r][c];
#pragma unroll
      for (int j = 0; j < 8; ++j) dst[j] = (bf16_t)src[j];
    }
    __syncthreads();

    Frag16 a;
    const bf16_t* ap = arow + k0 + khalf * 8;
    a.u[0] = *(const uint4*)ap;
    a.u[1] = *(const uint4*)(ap + 16);
#pragma unroll
    for (int nf = 0; nf < 4; ++nf) {
      Frag16 bfr;
      const bf16_t* bp = &wt[nf * 16 + l15][khalf * 8];
      bfr.u[0] = *(const uint4*)bp;
      bfr.u[1] = *(const uint4*)(bp + 16);
      acc[nf] = wmma_bf16(a.v, bfr.v, acc[nf]);
    }
  }

#pragma unroll
  for (int i = 0; i < 8; ++i) {
    const int m = mWave + i + ((lane & 16) ? 8 : 0);
#pragma unroll
    for (int nf = 0; nf < 4; ++nf)
      out[(size_t)m * DD + nBase + nf * 16 + l15] = acc[nf][i];
  }
}

// -------------------------------------------------------------------------
extern "C" void kernel_launch(void* const* d_in, const int* in_sizes, int n_in,
                              void* d_out, int out_size, void* d_ws, size_t ws_size,
                              hipStream_t stream) {
  const float* x  = (const float*)d_in[0];
  const float* qw = (const float*)d_in[1];
  const float* kw = (const float*)d_in[2];
  const float* vw = (const float*)d_in[3];
  const float* ow = (const float*)d_in[4];
  float* out = (float*)d_out;

  // workspace: Q, K, V^T, O each B*S*D bf16 = 8 MB -> 32 MB total
  const size_t elems = (size_t)BB * HH * SS * HD;
  bf16_t* qb  = (bf16_t*)d_ws;
  bf16_t* kb  = qb + elems;
  bf16_t* vtb = kb + elems;
  bf16_t* obuf = vtb + elems;

  dim3 blk(256);
  qkv_proj_kernel<<<dim3(DD / 64, MM / 128, 3), blk, 0, stream>>>(x, qw, kw, vw, qb, kb, vtb);
  attn_kernel<<<dim3(SS / 128, BB * HH), blk, 0, stream>>>(qb, kb, vtb, obuf);
  out_proj_kernel<<<dim3(DD / 64, MM / 128), blk, 0, stream>>>(obuf, ow, out);
}
